// RGLRU_30365418783093
// MI455X (gfx1250) — matrix operand order repaired and checked by reading, not compiled, not run on hardware
//
#include <hip/hip_runtime.h>

// ---------------- problem constants (from reference) ----------------
static constexpr int kB = 4;
static constexpr int kT = 4096;
static constexpr int kD = 2048;
static constexpr int Mtot = kB * kT;   // 16384 GEMM rows
static constexpr int Ntot = kD;        // 2048 GEMM cols
static constexpr int Ktot = kD;        // 2048 contraction

// ---------------- GEMM tiling ----------------
static constexpr int BM = 128, BN = 64, BK = 32;   // BK matches wmma 16x16x32
static constexpr int APITCH = BK + 8;              // u16 pitch, 80B rows (16B aligned, conflict-free)
static constexpr int WPITCH = BK + 8;

// ---------------- segmented scan ----------------
static constexpr int SEG    = 64;          // segments along T
static constexpr int SEGLEN = kT / SEG;    // 64 steps per segment

typedef __attribute__((ext_vector_type(16))) __bf16          v16bf;
typedef __attribute__((ext_vector_type(8)))  float           v8f;
typedef __attribute__((ext_vector_type(8)))  unsigned short  ushort8;

union FragBF { v16bf v; ushort8 h[2]; };

__device__ __forceinline__ unsigned short f2bf(float f) {
  unsigned int u = __float_as_uint(f);
  unsigned int r = u + 0x7FFFu + ((u >> 16) & 1u);   // round-to-nearest-even
  return (unsigned short)(r >> 16);
}
__device__ __forceinline__ float sigmoidf(float v) {
  return 1.0f / (1.0f + __expf(-v));
}

// =====================================================================
// Pass 0a: x (f32, [m][k]) -> bf16, same layout. Pure bandwidth.
// =====================================================================
__global__ __launch_bounds__(256)
void cvt_x_bf16(const float* __restrict__ x, unsigned short* __restrict__ xbf)
{
  const size_t i = ((size_t)blockIdx.x * 256 + threadIdx.x) * 8;
  const float4 a = *(const float4*)(x + i);
  const float4 b = *(const float4*)(x + i + 4);
  ushort8 o;
  o[0] = f2bf(a.x); o[1] = f2bf(a.y); o[2] = f2bf(a.z); o[3] = f2bf(a.w);
  o[4] = f2bf(b.x); o[5] = f2bf(b.y); o[6] = f2bf(b.z); o[7] = f2bf(b.w);
  *(ushort8*)(xbf + i) = o;
}

// =====================================================================
// Pass 0b: W (f32, [k][n]) -> bf16 transposed WT ([n][k]) via 32x32 LDS tile.
// B-fragments then read contiguous-K runs straight from memory order.
// =====================================================================
__global__ __launch_bounds__(256)
void cvt_w_transpose_bf16(const float* __restrict__ W, unsigned short* __restrict__ WT)
{
  __shared__ unsigned short tile[32][33];
  const int bn = blockIdx.x & 63;        // n tile
  const int bk = blockIdx.x >> 6;        // k tile
  const int n0 = bn * 32, k0 = bk * 32;
  const int t  = threadIdx.x;
  // read: 32 rows (k) x 32 cols (n), coalesced along n
  {
    const int kr = t >> 3;               // 0..31
    const int nc = (t & 7) * 4;          // 0..28
    const float4 v = *(const float4*)(W + (size_t)(k0 + kr) * Ntot + n0 + nc);
    tile[nc + 0][kr] = f2bf(v.x);
    tile[nc + 1][kr] = f2bf(v.y);
    tile[nc + 2][kr] = f2bf(v.z);
    tile[nc + 3][kr] = f2bf(v.w);
  }
  __syncthreads();
  // write: coalesced along k
  {
    const int nr = t >> 3;               // 0..31
    const int kc = (t & 7) * 4;          // 0..28
    unsigned short* dst = WT + (size_t)(n0 + nr) * Ktot + k0 + kc;
    dst[0] = tile[nr][kc + 0];
    dst[1] = tile[nr][kc + 1];
    dst[2] = tile[nr][kc + 2];
    dst[3] = tile[nr][kc + 3];
  }
}

// =====================================================================
// Kernel 1: dual-GEMM (rt = x@Wa, it = x@Wx) on bf16 inputs + fused gate
// epilogue. Register ping-pong overlaps next tile's global loads with WMMA.
// =====================================================================
__global__ __launch_bounds__(256)
void rglru_gates_gemm(const unsigned short* __restrict__ xbf,   // [m][k] bf16
                      const unsigned short* __restrict__ WaT,   // [n][k] bf16
                      const unsigned short* __restrict__ WxT,   // [n][k] bf16
                      const float* __restrict__ x,              // f32, for epilogue
                      const float* __restrict__ ba,
                      const float* __restrict__ bx,
                      const float* __restrict__ Lam,
                      float* __restrict__ at_ws,
                      float* __restrict__ g_ws)
{
  __shared__ __align__(16) unsigned short sA [BM * APITCH];  // x tile  [m][k]
  __shared__ __align__(16) unsigned short sWa[BN * WPITCH];  // Wa tile [n][k]
  __shared__ __align__(16) unsigned short sWx[BN * WPITCH];  // Wx tile [n][k]

  const int nBlocksN = Ntot / BN;
  const int m0 = (blockIdx.x / nBlocksN) * BM;
  const int n0 = (blockIdx.x % nBlocksN) * BN;

  const int tid  = threadIdx.x;
  const int wave = tid >> 5;      // 0..7, each owns a 16-row stripe
  const int lane = tid & 31;
  const int g    = lane >> 4;     // half-wave select
  const int l16  = lane & 15;

  // ---- per-thread tile-copy descriptors: 4 x 16B chunks per thread ----
  // chunks: [0..511]=A (128 rows x 4), [512..767]=Wa (64 x 4), [768..1023]=Wx
  const unsigned short* srcp[4];
  unsigned short*       dstp[4];
  #pragma unroll
  for (int j = 0; j < 4; ++j) {
    const int c = tid + 256 * j;
    if (c < 512) {
      const int row = c >> 2, cp = c & 3;
      srcp[j] = xbf + (size_t)(m0 + row) * Ktot + cp * 8;
      dstp[j] = &sA[row * APITCH + cp * 8];
    } else if (c < 768) {
      const int cc = c - 512, n = cc >> 2, cp = cc & 3;
      srcp[j] = WaT + (size_t)(n0 + n) * Ktot + cp * 8;
      dstp[j] = &sWa[n * WPITCH + cp * 8];
    } else {
      const int cc = c - 768, n = cc >> 2, cp = cc & 3;
      srcp[j] = WxT + (size_t)(n0 + n) * Ktot + cp * 8;
      dstp[j] = &sWx[n * WPITCH + cp * 8];
    }
  }

  v8f zero;
  #pragma unroll
  for (int i = 0; i < 8; ++i) zero[i] = 0.0f;
  v8f accR[4], accI[4];
  #pragma unroll
  for (int c = 0; c < 4; ++c) { accR[c] = zero; accI[c] = zero; }

  // prologue: stage tile k0=0 in registers
  ushort8 stage[4];
  #pragma unroll
  for (int j = 0; j < 4; ++j) stage[j] = *(const ushort8*)(srcp[j]);

  for (int k0 = 0; k0 < Ktot; k0 += BK) {
    // commit staged tile to LDS (ds_store_b128 x4)
    #pragma unroll
    for (int j = 0; j < 4; ++j) *(ushort8*)(dstp[j]) = stage[j];
    __syncthreads();

    // kick off next tile's global loads; they fly while WMMAs execute
    if (k0 + BK < Ktot) {
      #pragma unroll
      for (int j = 0; j < 4; ++j) stage[j] = *(const ushort8*)(srcp[j] + k0 + BK);
    }

    // ---- A fragment (16x32 bf16, lane holds contiguous K = g*16..g*16+15) ----
    FragBF fa;
    const int arow = wave * 16 + l16;
    fa.h[0] = *(const ushort8*)&sA[arow * APITCH + g * 16];
    fa.h[1] = *(const ushort8*)&sA[arow * APITCH + g * 16 + 8];

    #pragma unroll
    for (int c = 0; c < 4; ++c) {
      // B fragments (32x16 bf16, same K order as A: lane holds contiguous K = g*16..g*16+15)
      FragBF fba, fbx;
      const int col = c * 16 + l16;
      fba.h[0] = *(const ushort8*)&sWa[col * WPITCH + g * 16];
      fba.h[1] = *(const ushort8*)&sWa[col * WPITCH + g * 16 + 8];
      fbx.h[0] = *(const ushort8*)&sWx[col * WPITCH + g * 16];
      fbx.h[1] = *(const ushort8*)&sWx[col * WPITCH + g * 16 + 8];
      accR[c] = __builtin_amdgcn_wmma_f32_16x16x32_bf16(false, fa.v, false, fba.v,
                                                        (short)0, accR[c], false, false);
      accI[c] = __builtin_amdgcn_wmma_f32_16x16x32_bf16(false, fa.v, false, fbx.v,
                                                        (short)0, accI[c], false, false);
    }
    __syncthreads();
  }

  // ---- fused gate epilogue; C/D layout: VGPR r -> M = r + 8*(lane/16), N = lane%16 ----
  #pragma unroll
  for (int c = 0; c < 4; ++c) {
    const int nCol = n0 + c * 16 + l16;
    const float bav = ba[nCol];
    const float bxv = bx[nCol];
    const float ach = sigmoidf(Lam[nCol]);       // a = sigmoid(Lam)
    #pragma unroll
    for (int r = 0; r < 8; ++r) {
      const int mRow = m0 + wave * 16 + r + 8 * g;
      const float rt = sigmoidf(accR[c][r] + bav);
      const float it = sigmoidf(accI[c][r] + bxv);
      const float at = ach * exp2f(-3.0f * rt);  // a / 8^rt  (C = 8)
      const float xv = x[(size_t)mRow * Ktot + nCol];
      const float gt = sqrtf(fmaxf(1.0f - at * at, 0.0f)) * it * xv;
      const size_t idx = (size_t)mRow * Ntot + nCol;
      at_ws[idx] = at;
      g_ws[idx]  = gt;
    }
  }
}

// =====================================================================
// Segmented linear scan: h[t] = at[t]*h[t-1] + g[t]
// =====================================================================
__global__ __launch_bounds__(256)
void rglru_scan_pass1(const float* __restrict__ at_ws, const float* __restrict__ g_ws,
                      float* __restrict__ Aseg, float* __restrict__ Bseg)
{
  const int tid = blockIdx.x * 256 + threadIdx.x;   // < B*D*SEG
  const int d   = tid % kD;
  const int r   = tid / kD;
  const int b   = r % kB;
  const int seg = r / kB;
  size_t idx = ((size_t)(b * kT + seg * SEGLEN)) * kD + d;
  float A = 1.0f, Bc = 0.0f;
  #pragma unroll 4
  for (int i = 0; i < SEGLEN; ++i) {
    const float a_ = at_ws[idx];
    Bc = fmaf(a_, Bc, g_ws[idx]);
    A *= a_;
    idx += kD;
  }
  const size_t o = ((size_t)(seg * kB + b)) * kD + d;
  Aseg[o] = A; Bseg[o] = Bc;
}

__global__ __launch_bounds__(256)
void rglru_scan_pass2(const float* __restrict__ Aseg, const float* __restrict__ Bseg,
                      float* __restrict__ h0)
{
  const int tid = blockIdx.x * 256 + threadIdx.x;   // < B*D
  const int d = tid % kD;
  const int b = tid / kD;
  float h = 0.0f;
  for (int s = 0; s < SEG; ++s) {
    const size_t o = ((size_t)(s * kB + b)) * kD + d;
    h0[o] = h;                                      // h at segment start
    h = fmaf(Aseg[o], h, Bseg[o]);
  }
}

__global__ __launch_bounds__(256)
void rglru_scan_pass3(const float* __restrict__ at_ws, const float* __restrict__ g_ws,
                      const float* __restrict__ h0, float* __restrict__ out)
{
  const int tid = blockIdx.x * 256 + threadIdx.x;   // < B*D*SEG
  const int d   = tid % kD;
  const int r   = tid / kD;
  const int b   = r % kB;
  const int seg = r / kB;
  size_t idx = ((size_t)(b * kT + seg * SEGLEN)) * kD + d;
  float h = h0[((size_t)(seg * kB + b)) * kD + d];
  #pragma unroll 4
  for (int i = 0; i < SEGLEN; ++i) {
    const int t = seg * SEGLEN + i;
    h = fmaf(at_ws[idx], h, g_ws[idx]);
    out[((size_t)(t * kB + b)) * kD + d] = h;       // ys is (T, B, D)
    idx += kD;
  }
}

// =====================================================================
extern "C" void kernel_launch(void* const* d_in, const int* in_sizes, int n_in,
                              void* d_out, int out_size, void* d_ws, size_t ws_size,
                              hipStream_t stream)
{
  const float* x   = (const float*)d_in[0];
  const float* Wa  = (const float*)d_in[1];
  const float* Wx  = (const float*)d_in[2];
  const float* ba  = (const float*)d_in[3];
  const float* bx  = (const float*)d_in[4];
  const float* Lam = (const float*)d_in[5];
  float* out = (float*)d_out;

  float* ws = (float*)d_ws;
  const size_t MN   = (size_t)Mtot * Ntot;          // 33.5M
  const size_t KN   = (size_t)Ktot * Ntot;          // 4.2M
  const size_t SEGN = (size_t)SEG * kB * kD;        // 0.5M
  float* at_ws = ws;
  float* g_ws  = ws + MN;
  float* Aseg  = ws + 2 * MN;
  float* Bseg  = Aseg + SEGN;
  float* h0    = Bseg + SEGN;
  unsigned short* xbf = (unsigned short*)(h0 + SEGN);  // bf16 region
  unsigned short* WaT = xbf + MN;
  unsigned short* WxT = WaT + KN;

  dim3 blk(256);
  // precision prep (bandwidth-bound, one-time instead of per-tile)
  cvt_x_bf16<<<dim3((unsigned)(MN / (256 * 8))), blk, 0, stream>>>(x, xbf);
  cvt_w_transpose_bf16<<<dim3((Ntot / 32) * (Ktot / 32)), blk, 0, stream>>>(Wa, WaT);
  cvt_w_transpose_bf16<<<dim3((Ntot / 32) * (Ktot / 32)), blk, 0, stream>>>(Wx, WxT);
  // fused dual-GEMM + gates
  rglru_gates_gemm<<<dim3((Mtot / BM) * (Ntot / BN)), blk, 0, stream>>>(
      xbf, WaT, WxT, x, ba, bx, Lam, at_ws, g_ws);
  // segmented scan
  rglru_scan_pass1<<<dim3((kB * kD * SEG) / 256), blk, 0, stream>>>(at_ws, g_ws, Aseg, Bseg);
  rglru_scan_pass2<<<dim3((kB * kD) / 256), blk, 0, stream>>>(Aseg, Bseg, h0);
  rglru_scan_pass3<<<dim3((kB * kD * SEG) / 256), blk, 0, stream>>>(at_ws, g_ws, h0, out);
}
